// PatchGraphAttention_79946521247940
// MI455X (gfx1250) — matrix operand order repaired and checked by reading, not compiled, not run on hardware
//
#include <hip/hip_runtime.h>
#include <hip/hip_bf16.h>

typedef __bf16 v16bf __attribute__((ext_vector_type(16)));
typedef __bf16 bfx8  __attribute__((ext_vector_type(8)));
typedef float  v8f   __attribute__((ext_vector_type(8)));
typedef unsigned int u32x4 __attribute__((ext_vector_type(4)));
typedef int          i32x8 __attribute__((ext_vector_type(8)));
typedef int          i32x4 __attribute__((ext_vector_type(4)));

#define DMODEL 512
#define NHEAD  8
#define DHEAD  64
#define SEQ    2048
#define BATCH  4
#define TAU    300.0f
#define DELTA  300.0f

static __device__ __forceinline__ v8f zero8() {
  v8f z;
#pragma unroll
  for (int i = 0; i < 8; ++i) z[i] = 0.0f;
  return z;
}

static __device__ __forceinline__ v16bf make_v16(bfx8 lo, bfx8 hi) {
  v16bf r;
#pragma unroll
  for (int i = 0; i < 8; ++i) { r[i] = lo[i]; r[i + 8] = hi[i]; }
  return r;
}

static __device__ __forceinline__ bfx8 ld8(const __bf16* p) {
  return *reinterpret_cast<const bfx8*>(p);
}

// ---- Tensor Data Mover: 2-D tile (bf16 elements) global -> LDS -----------
// D# packing per CDNA5 ISA.
// This toolchain uses the 6-arg builtin: (g0, g1, g2, g3, g_extra, cpol).
static __device__ __forceinline__ void tdm_load_2d(unsigned lds_byte_off,
                                                   const void* gptr,
                                                   unsigned tensor_d0,
                                                   unsigned tensor_d1,
                                                   unsigned stride0,
                                                   unsigned tile_d0,
                                                   unsigned tile_d1) {
  unsigned long long ga = (unsigned long long)(uintptr_t)gptr;
  u32x4 g0;
  g0[0] = 1u;                                           // count=1, user mode
  g0[1] = lds_byte_off;                                 // lds_addr
  g0[2] = (unsigned)(ga & 0xffffffffull);               // global_addr[31:0]
  g0[3] = (unsigned)((ga >> 32) & 0x1ffffffull) | (2u << 30); // addr[56:32]|type=2
  i32x8 g1;
  g1[0] = (int)(1u << 16);                              // data_size=2B, mask=0
  g1[1] = (int)((tensor_d0 & 0xffffu) << 16);           // dim0[15:0]
  g1[2] = (int)((tensor_d0 >> 16) | ((tensor_d1 & 0xffffu) << 16));
  g1[3] = (int)((tensor_d1 >> 16) | (tile_d0 << 16));   // dim1[31:16] | tile0
  g1[4] = (int)(tile_d1 & 0xffffu);                     // tile1 | tile2=0
  g1[5] = (int)stride0;                                 // dim0_stride[31:0]
  g1[6] = 0;                                            // stride0 hi | stride1 lo
  g1[7] = 0;
  i32x4 z4 = {0, 0, 0, 0};
  i32x8 z8 = {0, 0, 0, 0, 0, 0, 0, 0};
  __builtin_amdgcn_tensor_load_to_lds(g0, g1, z4, z4, z8, 0);
}

static __device__ __forceinline__ unsigned lds_off(const void* p) {
  return (unsigned)(uintptr_t)p;   // generic->LDS: low 32 bits = LDS byte offset
}

// ---------------------------------------------------------------- convert
__global__ void cvt_f32_bf16(const float* __restrict__ src,
                             __bf16* __restrict__ dst, int n) {
  int i = blockIdx.x * blockDim.x + threadIdx.x;
  if (i < n) dst[i] = (__bf16)src[i];
}

// ---------------------------------------------------------------- QKV GEMM
// 16x64 output strip per wave: one A-frag reused for 4 WMMAs per k-step.
// All fragments loaded before the WMMA group so loadcnt waits stagger.
__global__ void __launch_bounds__(256)
qkv_proj_kernel(const __bf16* __restrict__ X,
                const __bf16* __restrict__ Wq, const __bf16* __restrict__ Wk,
                const __bf16* __restrict__ Wv,
                const float* __restrict__ bq, const float* __restrict__ bk,
                const float* __restrict__ bv,
                __bf16* __restrict__ Q, __bf16* __restrict__ Kd,
                __bf16* __restrict__ Vt) {
  const int wave = threadIdx.x >> 5;
  const int lane = threadIdx.x & 31;
  const int tile = blockIdx.x * 8 + wave;          // 3 * 512 * 8 strips
  const int sel  = tile / (512 * 8);
  const int rem  = tile - sel * (512 * 8);
  const int rt   = rem >> 3;                       // row tile (16 rows)
  const int st   = rem & 7;                        // 64-col strip == head id
  const int n    = lane & 15;
  const int hgrp = lane >> 4;
  const int h8 = hgrp * 8, h16 = hgrp * 16;

  const __bf16* W    = (sel == 0) ? Wq : (sel == 1) ? Wk : Wv;
  const float*  bias = (sel == 0) ? bq : (sel == 1) ? bk : bv;

  const __bf16* arow = X + (size_t)(rt * 16 + n) * DMODEL;
  const __bf16* brow[4];
#pragma unroll
  for (int j = 0; j < 4; ++j)
    brow[j] = W + (size_t)(st * 64 + j * 16 + n) * DMODEL + h16;

  v8f acc[4];
#pragma unroll
  for (int j = 0; j < 4; ++j) acc[j] = zero8();

#pragma unroll 2
  for (int kk = 0; kk < DMODEL; kk += 32) {
    const v16bf a = make_v16(ld8(arow + kk + h16), ld8(arow + kk + h16 + 8));
    v16bf b[4];
#pragma unroll
    for (int j = 0; j < 4; ++j)
      b[j] = make_v16(ld8(brow[j] + kk), ld8(brow[j] + kk + 8));
#pragma unroll
    for (int j = 0; j < 4; ++j)
      acc[j] = __builtin_amdgcn_wmma_f32_16x16x32_bf16(false, a, false, b[j],
                                                       (short)0, acc[j], false, false);
  }

#pragma unroll
  for (int j = 0; j < 4; ++j) {
    const int dh = j * 16 + n;                    // within head `st`
    const float be = bias[st * 64 + dh];
#pragma unroll
    for (int r = 0; r < 8; ++r) {                 // C/D: m = r + 8*(lane>>4)
      const int rowg = rt * 16 + r + h8;
      const int bb = rowg >> 11;
      const int mm = rowg & (SEQ - 1);
      const float val = acc[j][r] + be;
      if (sel == 0)
        Q[((size_t)(bb * NHEAD + st) * SEQ + mm) * DHEAD + dh] = (__bf16)val;
      else if (sel == 1)
        Kd[((size_t)(bb * NHEAD + st) * SEQ + mm) * DHEAD + dh] = (__bf16)val;
      else
        Vt[((size_t)(bb * NHEAD + st) * DHEAD + dh) * SEQ + mm] = (__bf16)val;
    }
  }
}

// ---------------------------------------------------------------- flash attention
// 4 waves/block; K/V chunks staged into LDS by the Tensor Data Mover,
// double-buffered, synchronized with s_wait_tensorcnt + barrier.
__global__ void __launch_bounds__(128)
flash_attn_kernel(const __bf16* __restrict__ Q, const __bf16* __restrict__ Kd,
                  const __bf16* __restrict__ Vt, const int* __restrict__ pmask,
                  __bf16* __restrict__ O) {
  __shared__ float biasTab[SEQ];                         // 8 KB
  __shared__ __align__(16) __bf16 kbuf[2][32 * DHEAD];   // 8 KB  [key][dh]
  __shared__ __align__(16) __bf16 vbuf[2][DHEAD * 32];   // 8 KB  [dh][key]
  __shared__ __align__(16) __bf16 pbuf[4][16 * 32];      // 4 KB

  const int bh   = blockIdx.x >> 5;   // b*8+h
  const int qblk = blockIdx.x & 31;
  const int b    = bh >> 3;
  const int hh   = bh & 7;
  const int w    = threadIdx.x >> 5;
  const int lane = threadIdx.x & 31;
  const int n    = lane & 15;
  const int hgrp = lane >> 4;
  const int h8 = hgrp * 8, h16 = hgrp * 16;

  for (int i = threadIdx.x; i < SEQ; i += blockDim.x)
    biasTab[i] = logf(expf(-(float)i * (DELTA / TAU)) + 1e-12f);

  const int qbase = qblk * 64 + w * 16;
  const __bf16* qrow = Q + ((size_t)bh * SEQ + qbase + n) * DHEAD;
  const v16bf qa0 = make_v16(ld8(qrow + h16),      ld8(qrow + h16 + 8));
  const v16bf qa1 = make_v16(ld8(qrow + 32 + h16), ld8(qrow + 40 + h16));

  const __bf16* Kb = Kd + (size_t)bh * SEQ * DHEAD;
  const __bf16* Vb = Vt + (size_t)bh * DHEAD * SEQ;
  const int*    pm = pmask + b * SEQ;

  float rmax[8], rsum[8];
  v8f acc[4];
#pragma unroll
  for (int r = 0; r < 8; ++r) { rmax[r] = -3.0e38f; rsum[r] = 0.0f; }
#pragma unroll
  for (int t = 0; t < 4; ++t) acc[t] = zero8();

  // prime double buffer: chunk 0 -> buf 0 (one wave issues per block)
  if (w == 0) {
    tdm_load_2d(lds_off(&kbuf[0][0]), Kb, SEQ * DHEAD, 1, SEQ * DHEAD, 32 * DHEAD, 1);
    tdm_load_2d(lds_off(&vbuf[0][0]), Vb, SEQ, DHEAD, SEQ, 32, DHEAD);
  }

  for (int key0 = 0; key0 < SEQ; key0 += 32) {
    const int cur = (key0 >> 5) & 1;
    if (w == 0) {
      if (key0 + 32 < SEQ) {
        tdm_load_2d(lds_off(&kbuf[cur ^ 1][0]), Kb + (size_t)(key0 + 32) * DHEAD,
                    SEQ * DHEAD, 1, SEQ * DHEAD, 32 * DHEAD, 1);
        tdm_load_2d(lds_off(&vbuf[cur ^ 1][0]), Vb + key0 + 32,
                    SEQ, DHEAD, SEQ, 32, DHEAD);
        __builtin_amdgcn_s_wait_tensorcnt(2);   // in-order: current 2 done
      } else {
        __builtin_amdgcn_s_wait_tensorcnt(0);
      }
    }
    __syncthreads();                            // K/V chunk visible to all waves

    // ---- S = Q K^T for two 16-key tiles (all K frags loaded up front)
    const __bf16* kc  = &kbuf[cur][0];
    const __bf16* kp0 = kc + (size_t)n * DHEAD + h16;
    const __bf16* kp1 = kc + (size_t)(16 + n) * DHEAD + h16;
    const v16bf kb00 = make_v16(ld8(kp0),      ld8(kp0 + 8));
    const v16bf kb01 = make_v16(ld8(kp0 + 32), ld8(kp0 + 40));
    const v16bf kb10 = make_v16(ld8(kp1),      ld8(kp1 + 8));
    const v16bf kb11 = make_v16(ld8(kp1 + 32), ld8(kp1 + 40));
    v8f z = zero8();
    v8f s0 = __builtin_amdgcn_wmma_f32_16x16x32_bf16(false, qa0, false, kb00,
                                                     (short)0, z, false, false);
    v8f s1 = __builtin_amdgcn_wmma_f32_16x16x32_bf16(false, qa0, false, kb10,
                                                     (short)0, z, false, false);
    s0 = __builtin_amdgcn_wmma_f32_16x16x32_bf16(false, qa1, false, kb01,
                                                 (short)0, s0, false, false);
    s1 = __builtin_amdgcn_wmma_f32_16x16x32_bf16(false, qa1, false, kb11,
                                                 (short)0, s1, false, false);

    const bool m0 = pm[key0 + n] != 0;
    const bool m1 = pm[key0 + 16 + n] != 0;

    float p0a[8], p1a[8];
#pragma unroll
    for (int r = 0; r < 8; ++r) {
      const int qi = qbase + r + h8;
      int d0 = qi - (key0 + n);      d0 = d0 < 0 ? -d0 : d0;
      int d1 = qi - (key0 + 16 + n); d1 = d1 < 0 ? -d1 : d1;
      float v0 = m0 ? (s0[r] * 0.125f + biasTab[d0]) : -1.0e9f;
      float v1 = m1 ? (s1[r] * 0.125f + biasTab[d1]) : -1.0e9f;
      float cm = fmaxf(v0, v1);
#pragma unroll
      for (int xm = 1; xm < 16; xm <<= 1) cm = fmaxf(cm, __shfl_xor(cm, xm, 16));
      const float nm   = fmaxf(rmax[r], cm);
      const float corr = __expf(rmax[r] - nm);
      rmax[r] = nm;
      const float p0 = __expf(v0 - nm);
      const float p1 = __expf(v1 - nm);
      float cs = p0 + p1;
#pragma unroll
      for (int xm = 1; xm < 16; xm <<= 1) cs += __shfl_xor(cs, xm, 16);
      rsum[r] = rsum[r] * corr + cs;
#pragma unroll
      for (int t = 0; t < 4; ++t) acc[t][r] *= corr;
      p0a[r] = p0; p1a[r] = p1;
    }

    // ---- restage P (C/D layout) -> A-fragment layout via LDS
#pragma unroll
    for (int r = 0; r < 8; ++r) {
      pbuf[w][(r + h8) * 32 + n]      = (__bf16)p0a[r];
      pbuf[w][(r + h8) * 32 + 16 + n] = (__bf16)p1a[r];
    }
    __syncthreads();
    const __bf16* prow = &pbuf[w][n * 32];
    const v16bf pa = make_v16(*(const bfx8*)(prow + h16),
                              *(const bfx8*)(prow + h16 + 8));
    v16bf vbf[4];
#pragma unroll
    for (int t = 0; t < 4; ++t) {
      const __bf16* vp = &vbuf[cur][(t * 16 + n) * 32] + h16;
      vbf[t] = make_v16(ld8(vp), ld8(vp + 8));
    }
#pragma unroll
    for (int t = 0; t < 4; ++t)
      acc[t] = __builtin_amdgcn_wmma_f32_16x16x32_bf16(false, pa, false, vbf[t],
                                                       (short)0, acc[t], false, false);
    __syncthreads();   // all reads of this buffer done before TDM overwrites
  }

  // ---- normalize and write attention output back to [b, m, h*64+d]
#pragma unroll
  for (int r = 0; r < 8; ++r) {
    const float inv = 1.0f / rsum[r];
    const int mrow = qbase + r + h8;
    __bf16* orow = O + ((size_t)b * SEQ + mrow) * DMODEL + hh * DHEAD;
#pragma unroll
    for (int t = 0; t < 4; ++t)
      orow[t * 16 + n] = (__bf16)(acc[t][r] * inv);
  }
}

// ---------------------------------------------------------------- output projection
__global__ void __launch_bounds__(256)
out_proj_kernel(const __bf16* __restrict__ O, const __bf16* __restrict__ Wo,
                const float* __restrict__ bo, const int* __restrict__ pmask,
                float* __restrict__ out) {
  const int wave = threadIdx.x >> 5;
  const int lane = threadIdx.x & 31;
  const int tile = blockIdx.x * 8 + wave;   // 512 * 8 strips
  const int rt = tile >> 3;
  const int st = tile & 7;
  const int n  = lane & 15;
  const int hgrp = lane >> 4;
  const int h8 = hgrp * 8, h16 = hgrp * 16;

  const __bf16* arow = O + (size_t)(rt * 16 + n) * DMODEL;
  const __bf16* brow[4];
#pragma unroll
  for (int j = 0; j < 4; ++j)
    brow[j] = Wo + (size_t)(st * 64 + j * 16 + n) * DMODEL + h16;

  v8f acc[4];
#pragma unroll
  for (int j = 0; j < 4; ++j) acc[j] = zero8();

#pragma unroll 2
  for (int kk = 0; kk < DMODEL; kk += 32) {
    const v16bf a = make_v16(ld8(arow + kk + h16), ld8(arow + kk + h16 + 8));
    v16bf b[4];
#pragma unroll
    for (int j = 0; j < 4; ++j)
      b[j] = make_v16(ld8(brow[j] + kk), ld8(brow[j] + kk + 8));
#pragma unroll
    for (int j = 0; j < 4; ++j)
      acc[j] = __builtin_amdgcn_wmma_f32_16x16x32_bf16(false, a, false, b[j],
                                                       (short)0, acc[j], false, false);
  }

#pragma unroll
  for (int j = 0; j < 4; ++j) {
    const int e = st * 64 + j * 16 + n;
    const float be = bo[e];
#pragma unroll
    for (int r = 0; r < 8; ++r) {
      const int rowg = rt * 16 + r + h8;     // flat b*SEQ + m
      const float mv = (pmask[rowg] != 0) ? 1.0f : 0.0f;
      out[(size_t)rowg * DMODEL + e] = (acc[j][r] + be) * mv;
    }
  }
}

// ---------------------------------------------------------------- launch
extern "C" void kernel_launch(void* const* d_in, const int* in_sizes, int n_in,
                              void* d_out, int out_size, void* d_ws, size_t ws_size,
                              hipStream_t stream) {
  (void)in_sizes; (void)n_in; (void)out_size; (void)ws_size;
  const float* h     = (const float*)d_in[0];
  const int*   pmask = (const int*)d_in[1];
  const float* Wq = (const float*)d_in[2];
  const float* bq = (const float*)d_in[3];
  const float* Wk = (const float*)d_in[4];
  const float* bk = (const float*)d_in[5];
  const float* Wv = (const float*)d_in[6];
  const float* bv = (const float*)d_in[7];
  const float* Wo = (const float*)d_in[8];
  const float* bo = (const float*)d_in[9];
  float* out = (float*)d_out;

  char* ws = (char*)d_ws;
  __bf16* Xbf  = (__bf16*)(ws);                 //  8 MB  h as bf16
  __bf16* Wqbf = (__bf16*)(ws +  8388608);      // 512 KB each
  __bf16* Wkbf = (__bf16*)(ws +  8912896);
  __bf16* Wvbf = (__bf16*)(ws +  9437184);
  __bf16* Wobf = (__bf16*)(ws +  9961472);
  __bf16* Qbf  = (__bf16*)(ws + 10485760);      //  8 MB  [b,h,m,dh]
  __bf16* Kbf  = (__bf16*)(ws + 18874368);      //  8 MB  [b,h,m,dh]
  __bf16* Vtbf = (__bf16*)(ws + 27262976);      //  8 MB  [b,h,dh,m]
  __bf16* Obf  = (__bf16*)(ws + 35651584);      //  8 MB  [b,m,512]

  const int NH = BATCH * SEQ * DMODEL;          // 4194304
  const int NW = DMODEL * DMODEL;               // 262144
  cvt_f32_bf16<<<(NH + 255) / 256, 256, 0, stream>>>(h,  Xbf,  NH);
  cvt_f32_bf16<<<(NW + 255) / 256, 256, 0, stream>>>(Wq, Wqbf, NW);
  cvt_f32_bf16<<<(NW + 255) / 256, 256, 0, stream>>>(Wk, Wkbf, NW);
  cvt_f32_bf16<<<(NW + 255) / 256, 256, 0, stream>>>(Wv, Wvbf, NW);
  cvt_f32_bf16<<<(NW + 255) / 256, 256, 0, stream>>>(Wo, Wobf, NW);

  qkv_proj_kernel<<<1536, 256, 0, stream>>>(Xbf, Wqbf, Wkbf, Wvbf,
                                            bq, bk, bv, Qbf, Kbf, Vtbf);
  flash_attn_kernel<<<1024, 128, 0, stream>>>(Qbf, Kbf, Vtbf, pmask, Obf);
  out_proj_kernel<<<512, 256, 0, stream>>>(Obf, Wobf, bo, pmask, out);
}
